// Attention_23476291240422
// MI455X (gfx1250) — matrix operand order repaired and checked by reading, not compiled, not run on hardware
//
#include <hip/hip_runtime.h>

// ---------------------------------------------------------------------------
// Fused attention block for MI455X (gfx1250, wave32, WMMA).
//   out = Attention(x @ w_qkv) @ w_out + b_out
// f16 compute / f32 accumulate via v_wmma_f32_16x16x32_f16.
// Staging uses GLOBAL_LOAD_ASYNC_TO_LDS (ASYNCcnt) when the builtins exist.
//
// Operand layouts as used here (wave32):
//   A 16x32 f16 : lane = row; lanes 0-15 hold K 0..15 contiguous,
//                 lanes 16-31 hold K 16..31 contiguous   -> make_afrag(p + half*16)
//   B 32x16 f16 : lane&15 = col; lanes 0-15 hold K 0..15 contiguous,
//                 lanes 16-31 hold K 16..31 contiguous   -> one v16h at +half*16
//   C/D 16x16 f32: lane&15 = N; VGPR r holds M = r + 8*(lane>=16)
// ---------------------------------------------------------------------------

typedef __attribute__((ext_vector_type(16))) _Float16 v16h;
typedef __attribute__((ext_vector_type(8)))  _Float16 v8h;
typedef __attribute__((ext_vector_type(4)))  _Float16 v4h;
typedef __attribute__((ext_vector_type(8)))  float    v8f;
typedef __attribute__((ext_vector_type(4)))  float    v4f;
typedef __attribute__((ext_vector_type(4)))  int      v4i;

#define HEADS      16
#define DIM_HEAD   128
#define NONCAUSAL  128
#define BATCH      4
#define SEQ        2048
#define DIM        2048
#define INNER      2048   // HEADS * DIM_HEAD
#define NQKV       6144   // 3 * INNER
#define MROWS      8192   // BATCH * SEQ

// ---- async direct-to-LDS staging (gfx1250), with sync fallback -------------
#if defined(__has_builtin)
#if __has_builtin(__builtin_amdgcn_global_load_async_to_lds_b128) && \
    __has_builtin(__builtin_amdgcn_s_wait_asynccnt)
#define USE_ASYNC_LDS 1
#endif
#endif

__device__ inline void copy16_to_lds(_Float16* dst_lds, const _Float16* src) {
#ifdef USE_ASYNC_LDS
  __builtin_amdgcn_global_load_async_to_lds_b128(
      (__attribute__((address_space(1))) v4i*)src,
      (__attribute__((address_space(3))) v4i*)dst_lds, 0, 0);
#else
  *(v8h*)dst_lds = *(const v8h*)src;
#endif
}

__device__ inline void async_stage_wait() {
#ifdef USE_ASYNC_LDS
  __builtin_amdgcn_s_wait_asynccnt(0);
#endif
}

// A-operand fragment (16x32): per lane, 16 contiguous halves at p0.
__device__ inline v16h make_afrag(const _Float16* p0) {
  v8h lo = *(const v8h*)p0;
  v8h up = *(const v8h*)(p0 + 8);
  v16h r;
#pragma unroll
  for (int i = 0; i < 8; ++i) { r[i] = lo[i]; r[8 + i] = up[i]; }
  return r;
}

// B-operand fragment (32x16): per lane, 16 contiguous halves (32B aligned).
__device__ inline v16h make_bfrag(const _Float16* p0) {
  return *(const v16h*)p0;
}

// ---------------------------------------------------------------------------
// Vectorized f32 -> f16 conversion (grid-stride, 4 elems/thread/step).
// ---------------------------------------------------------------------------
__global__ void f32_to_f16_kernel(const float* __restrict__ in,
                                  _Float16* __restrict__ out, int n4) {
  int i = blockIdx.x * blockDim.x + threadIdx.x;
  int stride = gridDim.x * blockDim.x;
  for (; i < n4; i += stride) {
    v4f v = ((const v4f*)in)[i];
    v4h h;
#pragma unroll
    for (int j = 0; j < 4; ++j) h[j] = (_Float16)v[j];
    ((v4h*)out)[i] = h;
  }
}

// ---------------------------------------------------------------------------
// Tiled WMMA GEMM: C[M,N] = A[M,K] @ B[K,N], f16 in, f32 accum.
// Block tile 128x256, BK=32, 256 threads = 8 waves (2x4), wave tile 64x64.
// Double-buffered LDS; A staged async-to-LDS; B staged transposed ([n][k])
// with 4-row-paired b64 stores.
// QKV_MODE: scatter f16 results into q/k/v [b,h,n,d] tensors.
// else:     store f32 + bias into outF.
// ---------------------------------------------------------------------------
template <bool QKV_MODE>
__global__ __launch_bounds__(256)
void gemm_wmma_kernel(const _Float16* __restrict__ A,
                      const _Float16* __restrict__ B,
                      int M, int N, int K,
                      _Float16* __restrict__ qd, _Float16* __restrict__ kd,
                      _Float16* __restrict__ vd,
                      float* __restrict__ outF, const float* __restrict__ bias) {
  __shared__ __align__(16) _Float16 As[2 * 128 * 32];  // [buf][m][k]
  __shared__ __align__(16) _Float16 Bs[2 * 256 * 32];  // [buf][n][k] transposed

  const int tid   = threadIdx.x;
  const int lane  = tid & 31;
  const int wid   = tid >> 5;      // 0..7
  const int wm    = wid >> 2;      // 0..1 : 64 rows per wave
  const int wn    = wid & 3;       // 0..3 : 64 cols per wave
  const int lh    = lane & 15;
  const int half  = lane >> 4;     // 0 or 1

  const int m0 = blockIdx.y * 128;
  const int n0 = blockIdx.x * 256;

  v8f zero = {};
  v8f acc[4][4];
#pragma unroll
  for (int i = 0; i < 4; ++i)
#pragma unroll
    for (int j = 0; j < 4; ++j) acc[i][j] = zero;

  // --- staging helpers (each thread moves fixed chunks) ---
  auto stageA = [&](int k0, int buf) {
    _Float16* Asb = &As[buf * 128 * 32];
#pragma unroll
    for (int it = 0; it < 2; ++it) {
      int chunk = tid + it * 256;        // 0..511
      int row = chunk >> 2;
      int cc  = chunk & 3;
      copy16_to_lds(&Asb[row * 32 + cc * 8],
                    &A[(size_t)(m0 + row) * K + k0 + cc * 8]);
    }
  };
  auto stageB = [&](int k0, int buf) {
    _Float16* Bsb = &Bs[buf * 256 * 32];
    // Each thread: 4 consecutive k-rows x 8 n-cols -> 4 global b128 loads,
    // 8 b64 LDS stores (k-contiguous quads in the [n][k] layout).
    int kq = (tid >> 5) * 4;             // 0,4,...,28
    int nc = tid & 31;                   // n-group of 8
    const _Float16* gp = &B[(size_t)(k0 + kq) * N + n0 + nc * 8];
    v8h r0 = *(const v8h*)(gp);
    v8h r1 = *(const v8h*)(gp + (size_t)N);
    v8h r2 = *(const v8h*)(gp + (size_t)2 * N);
    v8h r3 = *(const v8h*)(gp + (size_t)3 * N);
#pragma unroll
    for (int e = 0; e < 8; ++e) {
      v4h q4 = {r0[e], r1[e], r2[e], r3[e]};
      *(v4h*)&Bsb[(nc * 8 + e) * 32 + kq] = q4;
    }
  };

  stageA(0, 0);
  stageB(0, 0);
  async_stage_wait();
  __syncthreads();

  for (int k0 = 0; k0 < K; k0 += 32) {
    const int cur = (k0 >> 5) & 1;
    if (k0 + 32 < K) {                    // prefetch next tile into other buf
      stageA(k0 + 32, cur ^ 1);
      stageB(k0 + 32, cur ^ 1);
    }

    const _Float16* Asb = &As[cur * 128 * 32];
    const _Float16* Bsb = &Bs[cur * 256 * 32];
    v16h afr[4], bfr[4];
#pragma unroll
    for (int i = 0; i < 4; ++i)
      afr[i] = make_afrag(&Asb[(wm * 64 + i * 16 + lh) * 32 + half * 16]);
#pragma unroll
    for (int j = 0; j < 4; ++j)
      bfr[j] = make_bfrag(&Bsb[(wn * 64 + j * 16 + lh) * 32 + half * 16]);
#pragma unroll
    for (int i = 0; i < 4; ++i)
#pragma unroll
      for (int j = 0; j < 4; ++j)
        acc[i][j] = __builtin_amdgcn_wmma_f32_16x16x32_f16(
            false, afr[i], false, bfr[j], (short)0, acc[i][j], false, false);

    async_stage_wait();
    __syncthreads();
  }

  // Epilogue. C layout: lane L holds N = L&15, M = r + 8*(L>=16).
#pragma unroll
  for (int i = 0; i < 4; ++i) {
#pragma unroll
    for (int j = 0; j < 4; ++j) {
#pragma unroll
      for (int r = 0; r < 8; ++r) {
        int mrow = m0 + wm * 64 + i * 16 + r + half * 8;
        int ncol = n0 + wn * 64 + j * 16 + lh;
        float val = acc[i][j][r];
        if (QKV_MODE) {
          int sec  = ncol >> 11;        // /2048 : 0=q 1=k 2=v
          int c2   = ncol & 2047;
          int head = c2 >> 7;
          int dd   = c2 & 127;
          int bi   = mrow >> 11;
          int ni   = mrow & 2047;
          _Float16* dst = (sec == 0) ? qd : (sec == 1) ? kd : vd;
          dst[(((size_t)(bi * HEADS + head) * SEQ + ni) << 7) + dd] =
              (_Float16)val;
        } else {
          outF[(size_t)mrow * N + ncol] = val + bias[ncol];
        }
      }
    }
  }
}

// ---------------------------------------------------------------------------
// Flash attention: one block per (b*h, 128-row q-tile). 8 waves x 16 rows.
// Key tile of 64; online softmax; causal mask with 128x128 prefix unmasked.
// ---------------------------------------------------------------------------
__global__ __launch_bounds__(256)
void flash_attn_kernel(const _Float16* __restrict__ Q,
                       const _Float16* __restrict__ Kmat,
                       const _Float16* __restrict__ V,
                       _Float16* __restrict__ Attn) {
  __shared__ __align__(16) _Float16 Ks[64 * 128];      // [key][d]
  __shared__ __align__(16) _Float16 Vs[128 * 64];      // [d][key] (transposed)
  __shared__ __align__(16) _Float16 Ps[8 * 16 * 64];   // per-wave P staging

  const int tid  = threadIdx.x;
  const int lane = tid & 31;
  const int w    = tid >> 5;
  const int lh   = lane & 15;
  const int half = lane >> 4;

  const int bh = blockIdx.y;            // 0..63
  const int qt = blockIdx.x;            // 0..15
  const int q0 = qt * 128;
  const float scale = 0.08838834764831845f;  // 1/sqrt(128)

  const _Float16* Qb = Q    + (size_t)bh * SEQ * DIM_HEAD;
  const _Float16* Kb = Kmat + (size_t)bh * SEQ * DIM_HEAD;
  const _Float16* Vb = V    + (size_t)bh * SEQ * DIM_HEAD;

  // Q fragments for this wave's 16 rows (d split into 4 chunks of 32).
  v16h qf[4];
  {
    const _Float16* base = Qb + (size_t)(q0 + w * 16 + lh) * DIM_HEAD;
#pragma unroll
    for (int c = 0; c < 4; ++c)
      qf[c] = make_afrag(base + c * 32 + half * 16);
  }

  v8f zero = {};
  float mrun[8], lrun[8];
  v8f o[8];
#pragma unroll
  for (int r = 0; r < 8; ++r) { mrun[r] = -3.0e38f; lrun[r] = 0.0f; }
#pragma unroll
  for (int c = 0; c < 8; ++c) o[c] = zero;

  const int ntiles = (qt + 1) * 2;      // keys up to q0+127 (64 per tile)
  for (int jt = 0; jt < ntiles; ++jt) {
    const int j0 = jt * 64;

    // Stage K tile 64x128 row-major (async direct-to-LDS when available).
#pragma unroll
    for (int it = 0; it < 4; ++it) {
      int chunk = tid + it * 256;       // 0..1023
      int kr = chunk >> 4;              // key 0..63
      int dc = chunk & 15;
      copy16_to_lds(&Ks[kr * 128 + dc * 8],
                    &Kb[(size_t)(j0 + kr) * 128 + dc * 8]);
    }
    // Stage V tile transposed: Vs[d][key], 4-key quads -> b64 stores.
    {
      int kb = (tid >> 4) * 4;          // key base 0,4,...,60
      int dg = tid & 15;                // d-group of 8
      const _Float16* gp = &Vb[(size_t)(j0 + kb) * 128 + dg * 8];
      v8h r0 = *(const v8h*)(gp);
      v8h r1 = *(const v8h*)(gp + 128);
      v8h r2 = *(const v8h*)(gp + 256);
      v8h r3 = *(const v8h*)(gp + 384);
#pragma unroll
      for (int e = 0; e < 8; ++e) {
        v4h q4 = {r0[e], r1[e], r2[e], r3[e]};
        *(v4h*)&Vs[(dg * 8 + e) * 64 + kb] = q4;
      }
    }
    async_stage_wait();
    __syncthreads();

    // S = Q . K^T -> four 16x16 f32 tiles (64 keys).
    v8f s[4];
#pragma unroll
    for (int nh = 0; nh < 4; ++nh) {
      v8f a = zero;
#pragma unroll
      for (int c = 0; c < 4; ++c) {
        v16h kf = make_bfrag(&Ks[(nh * 16 + lh) * 128 + c * 32 + half * 16]);
        a = __builtin_amdgcn_wmma_f32_16x16x32_f16(
            false, qf[c], false, kf, (short)0, a, false, false);
      }
      s[nh] = a;
    }

    // Scale + causal mask (prefix 128x128 unmasked).
#pragma unroll
    for (int nh = 0; nh < 4; ++nh) {
#pragma unroll
      for (int r = 0; r < 8; ++r) {
        int row = q0 + w * 16 + r + half * 8;
        int col = j0 + nh * 16 + lh;
        float sv = s[nh][r] * scale;
        bool masked = (col > row) && !((row < NONCAUSAL) && (col < NONCAUSAL));
        s[nh][r] = masked ? -3.0e38f : sv;
      }
    }

    // Online softmax per row: butterfly over the 16-lane half-wave.
    float corr[8];
#pragma unroll
    for (int r = 0; r < 8; ++r) {
      float mx = fmaxf(fmaxf(s[0][r], s[1][r]), fmaxf(s[2][r], s[3][r]));
#pragma unroll
      for (int off = 1; off < 16; off <<= 1)
        mx = fmaxf(mx, __shfl_xor(mx, off, 32));
      float mnew = fmaxf(mrun[r], mx);
      float cr = __expf(mrun[r] - mnew);
      float p0 = __expf(s[0][r] - mnew);
      float p1 = __expf(s[1][r] - mnew);
      float p2 = __expf(s[2][r] - mnew);
      float p3 = __expf(s[3][r] - mnew);
      s[0][r] = p0; s[1][r] = p1; s[2][r] = p2; s[3][r] = p3;
      float sum = (p0 + p1) + (p2 + p3);
#pragma unroll
      for (int off = 1; off < 16; off <<= 1)
        sum += __shfl_xor(sum, off, 32);
      lrun[r] = lrun[r] * cr + sum;
      mrun[r] = mnew;
      corr[r] = cr;
    }
#pragma unroll
    for (int c = 0; c < 8; ++c)
#pragma unroll
      for (int r = 0; r < 8; ++r) o[c][r] *= corr[r];

    // C-layout P -> A-layout fragments via per-wave LDS staging (f16).
    _Float16* Pw = &Ps[w * 16 * 64];
#pragma unroll
    for (int nh = 0; nh < 4; ++nh)
#pragma unroll
      for (int r = 0; r < 8; ++r)
        Pw[(r + half * 8) * 64 + nh * 16 + lh] = (_Float16)s[nh][r];
    v16h pf[2];
#pragma unroll
    for (int t = 0; t < 2; ++t)
      pf[t] = make_afrag(&Pw[lh * 64 + t * 32 + half * 16]);

    // O += P . V   (8 d-chunks of 16, two 32-key chunks)
#pragma unroll
    for (int c = 0; c < 8; ++c) {
#pragma unroll
      for (int t = 0; t < 2; ++t) {
        v16h vf = make_bfrag(&Vs[(c * 16 + lh) * 64 + t * 32 + half * 16]);
        o[c] = __builtin_amdgcn_wmma_f32_16x16x32_f16(
            false, pf[t], false, vf, (short)0, o[c], false, false);
      }
    }
    __syncthreads();
  }

  // Normalize and store to attn_out [b, n, h*d] as f16.
  const int b = bh >> 4;
  const int h = bh & 15;
#pragma unroll
  for (int r = 0; r < 8; ++r) {
    float inv = 1.0f / lrun[r];
    int row = q0 + w * 16 + r + half * 8;
    size_t base = ((size_t)b * SEQ + row) * INNER + h * DIM_HEAD;
#pragma unroll
    for (int c = 0; c < 8; ++c)
      Attn[base + c * 16 + lh] = (_Float16)(o[c][r] * inv);
  }
}

// ---------------------------------------------------------------------------
// Launch: convert -> qkv gemm -> flash attention -> out gemm (+bias).
// ---------------------------------------------------------------------------
extern "C" void kernel_launch(void* const* d_in, const int* in_sizes, int n_in,
                              void* d_out, int out_size, void* d_ws,
                              size_t ws_size, hipStream_t stream) {
  const float* x     = (const float*)d_in[0];
  const float* w_qkv = (const float*)d_in[1];
  const float* w_out = (const float*)d_in[2];
  const float* b_out = (const float*)d_in[3];
  float* out = (float*)d_out;

  char* ws = (char*)d_ws;
  size_t off = 0;
  auto alloc = [&](size_t bytes) -> char* {
    char* p = ws + off;
    off += (bytes + 255) & ~(size_t)255;
    return p;
  };
  _Float16* x_h  = (_Float16*)alloc((size_t)MROWS * DIM * 2);
  _Float16* wq_h = (_Float16*)alloc((size_t)DIM * NQKV * 2);
  _Float16* wo_h = (_Float16*)alloc((size_t)INNER * DIM * 2);
  _Float16* q_h  = (_Float16*)alloc((size_t)BATCH * HEADS * SEQ * DIM_HEAD * 2);
  _Float16* k_h  = (_Float16*)alloc((size_t)BATCH * HEADS * SEQ * DIM_HEAD * 2);
  _Float16* v_h  = (_Float16*)alloc((size_t)BATCH * HEADS * SEQ * DIM_HEAD * 2);
  _Float16* a_h  = (_Float16*)alloc((size_t)MROWS * INNER * 2);

  f32_to_f16_kernel<<<dim3(1024), dim3(256), 0, stream>>>(x, x_h,
                                                          MROWS * DIM / 4);
  f32_to_f16_kernel<<<dim3(512), dim3(256), 0, stream>>>(w_qkv, wq_h,
                                                         DIM * NQKV / 4);
  f32_to_f16_kernel<<<dim3(256), dim3(256), 0, stream>>>(w_out, wo_h,
                                                         INNER * DIM / 4);

  // QKV projection: [8192,2048] @ [2048,6144] -> q/k/v [b,h,n,d]
  gemm_wmma_kernel<true><<<dim3(NQKV / 256, MROWS / 128), dim3(256), 0,
                           stream>>>(x_h, wq_h, MROWS, NQKV, DIM, q_h, k_h,
                                     v_h, (float*)nullptr,
                                     (const float*)nullptr);

  // Flash attention: grid (q-tiles, b*h)
  flash_attn_kernel<<<dim3(SEQ / 128, BATCH * HEADS), dim3(256), 0, stream>>>(
      q_h, k_h, v_h, a_h);

  // Output projection + bias: [8192,2048] @ [2048,2048] -> f32 out
  gemm_wmma_kernel<false><<<dim3(DIM / 256, MROWS / 128), dim3(256), 0,
                            stream>>>(a_h, wo_h, MROWS, DIM, DIM,
                                      (_Float16*)nullptr, (_Float16*)nullptr,
                                      (_Float16*)nullptr, out, b_out);
}
